// NTN_25357486916247
// MI455X (gfx1250) — compile-verified, matrix operand order checked by reading, not run on hardware
//
#include <hip/hip_runtime.h>
#include <hip/hip_bf16.h>

// Problem constants (from the reference)
#define Bdim 64
#define Qdim 256
#define Adim 256
#define Ddim 256
#define Kdim 16
#define QT   64    // q-rows per workgroup
#define DC   32    // reduction chunk (WMMA K per step)
#define QP   264   // padded row stride for qT_lds (528B: lane stride 16B mod 256)
#define CP   40    // padded row stride for chunk_lds (80B: 16B-aligned, spreads banks)

typedef __attribute__((ext_vector_type(16))) __bf16 v16bf;
typedef __attribute__((ext_vector_type(8)))  float  v8f;
typedef __attribute__((ext_vector_type(8)))  __bf16 bf16x8;
typedef __attribute__((ext_vector_type(4)))  __bf16 bf16x4;
typedef __attribute__((ext_vector_type(2)))  __bf16 bf16x2;

__device__ __forceinline__ bf16x2 cvt2(float x, float y) {
  bf16x2 r; r[0] = (__bf16)x; r[1] = (__bf16)y; return r;
}
__device__ __forceinline__ bf16x4 cvt4(float4 a) {
  bf16x4 r;
  r[0] = (__bf16)a.x; r[1] = (__bf16)a.y; r[2] = (__bf16)a.z; r[3] = (__bf16)a.w;
  return r;
}
__device__ __forceinline__ bf16x8 cvt8(float4 a, float4 b) {
  bf16x8 r;
  r[0] = (__bf16)a.x; r[1] = (__bf16)a.y; r[2] = (__bf16)a.z; r[3] = (__bf16)a.w;
  r[4] = (__bf16)b.x; r[5] = (__bf16)b.y; r[6] = (__bf16)b.z; r[7] = (__bf16)b.w;
  return r;
}

// Build a 16-element bf16 fragment from two contiguous 16B LDS chunks.
__device__ __forceinline__ v16bf mkfrag(const __bf16* p0, const __bf16* p1) {
  bf16x8 lo = *(const bf16x8*)p0;
  bf16x8 hi = *(const bf16x8*)p1;
  v16bf f;
#pragma unroll
  for (int i = 0; i < 8; ++i) { f[i] = lo[i]; f[i + 8] = hi[i]; }
  return f;
}

__device__ __forceinline__ float sigmoidf(float x) {
  return 1.0f / (1.0f + __expf(-x));
}

__global__ __launch_bounds__(256)
void ntn_fused_wmma(const float* __restrict__ q_em,   // (B,Q,D)
                    const float* __restrict__ da_em,  // (B,A,D)
                    const float* __restrict__ w,      // (K,D,D)
                    const float* __restrict__ V,      // (K,2D)
                    const float* __restrict__ bias,   // (K,1,1)
                    float* __restrict__ out)          // (B,K,Q,A)
{
  // qT_lds holds the Q-tile [q][d] (bf16) in phase 1, then T [q][e] in phase 2.
  __shared__ __bf16 qT_lds[QT * QP];         // ~33 KB, padded rows
  // chunk_lds holds a transposed w chunk [e][d] (phase 1) or a da chunk [a][e] (phase 2).
  __shared__ __bf16 chunk_lds[Adim * CP];    // ~20 KB, padded rows
  __shared__ float  vq_s[QT];
  __shared__ float  vd_s[Adim];

  const int tid  = threadIdx.x;
  const int lane = tid & 31;
  const int wv   = tid >> 5;          // 0..7 (8 waves, wave32)
  const int hif  = lane >> 4;         // 0/1 : K-half / M-half owned by this lane
  const int ln   = lane & 15;         // 0..15 : M (A-frag) or N (B-frag) index

  const int qt    = blockIdx.x & 3;           // q-tile
  const int k     = (blockIdx.x >> 2) & 15;   // filter
  const int b     = blockIdx.x >> 6;          // batch
  const int qbase = qt * QT;

  const int mtile0 = wv * 2;   // this wave owns M-tiles {mtile0, mtile0+1} x N-tiles {0..3}

  // ---- stage Q tile into LDS (bf16), float4 loads + 8B packed stores ----
  const float* qrow0 = q_em + ((size_t)b * Qdim + qbase) * Ddim;
#pragma unroll
  for (int i = 0; i < (QT * Ddim) / (4 * 256); ++i) {   // 16 float4 quads/thread
    int fq   = i * 256 + tid;
    int row  = fq >> 6;                 // 64 quads per 256-wide row
    int col4 = (fq & 63) * 4;
    const float4 v = *(const float4*)(qrow0 + (size_t)row * Ddim + col4);
    *(bf16x4*)&qT_lds[row * QP + col4] = cvt4(v);
  }

  // ---- small VALU dots: vq (64 rows) and vd (256 rows) ----
  {
    const float4* Vq4 = (const float4*)(V + (size_t)k * 2 * Ddim);
    const float4* Vd4 = (const float4*)(V + (size_t)k * 2 * Ddim + Ddim);
    if (tid < QT) {
      const float4* qr = (const float4*)(qrow0 + (size_t)tid * Ddim);
      float s = 0.f;
      for (int d = 0; d < Ddim / 4; ++d) {
        float4 a = Vq4[d], c = qr[d];
        s += a.x * c.x + a.y * c.y + a.z * c.z + a.w * c.w;
      }
      vq_s[tid] = s;
    }
    const float4* dr = (const float4*)(da_em + ((size_t)b * Adim + tid) * Ddim);
    float s = 0.f;
    for (int d = 0; d < Ddim / 4; ++d) {
      float4 a = Vd4[d], c = dr[d];
      s += a.x * c.x + a.y * c.y + a.z * c.z + a.w * c.w;
    }
    vd_s[tid] = s;
  }

  const v8f vzero = {};

  // =======================================================================
  // Phase 1 (transposed): Tt[e][q] = sum_d wT[e][d] * Q[q][d]
  //   A = wT chunk [e][d] (M = e), B = Q-tile [q][d] (N = q)
  // =======================================================================
  v8f acc[2][4];
#pragma unroll
  for (int mt = 0; mt < 2; ++mt)
#pragma unroll
    for (int nt = 0; nt < 4; ++nt) acc[mt][nt] = vzero;

  const float* wk = w + (size_t)k * Ddim * Ddim;

  for (int dc = 0; dc < Ddim; dc += DC) {
    // stage w[dc..dc+31][0..255] transposed into chunk_lds[e][d]:
    // each thread moves a 2(d) x 4(e) patch via two b128 loads -> 4 b32 stores.
#pragma unroll
    for (int i = 0; i < (DC * Ddim) / (8 * 256); ++i) {   // 4 iters
      int flat = i * 256 + tid;
      int e4 = (flat & 63) * 4;   // 0..252
      int dp = flat >> 6;         // 0..15 : d-pair
      const float4 r0 = *(const float4*)(wk + (size_t)(dc + 2 * dp)     * Ddim + e4);
      const float4 r1 = *(const float4*)(wk + (size_t)(dc + 2 * dp + 1) * Ddim + e4);
      *(bf16x2*)&chunk_lds[(e4 + 0) * CP + 2 * dp] = cvt2(r0.x, r1.x);
      *(bf16x2*)&chunk_lds[(e4 + 1) * CP + 2 * dp] = cvt2(r0.y, r1.y);
      *(bf16x2*)&chunk_lds[(e4 + 2) * CP + 2 * dp] = cvt2(r0.z, r1.z);
      *(bf16x2*)&chunk_lds[(e4 + 3) * CP + 2 * dp] = cvt2(r0.w, r1.w);
    }
    __syncthreads();   // chunk ready (covers qT_lds / vq / vd on iter 0)

    v16bf afrag[2];
#pragma unroll
    for (int mt = 0; mt < 2; ++mt) {
      int e = (mtile0 + mt) * 16 + ln;
      afrag[mt] = mkfrag(&chunk_lds[e * CP + hif * 8],
                         &chunk_lds[e * CP + hif * 8 + 16]);
    }
    v16bf bfrag[4];
#pragma unroll
    for (int nt = 0; nt < 4; ++nt) {
      int q = nt * 16 + ln;
      bfrag[nt] = mkfrag(&qT_lds[q * QP + dc + hif * 8],
                         &qT_lds[q * QP + dc + hif * 8 + 16]);
    }
#pragma unroll
    for (int mt = 0; mt < 2; ++mt)
#pragma unroll
      for (int nt = 0; nt < 4; ++nt)
        acc[mt][nt] = __builtin_amdgcn_wmma_f32_16x16x32_bf16(
            false, afrag[mt], false, bfrag[nt], (short)0, acc[mt][nt], false, false);
    __syncthreads();   // everyone done reading before chunk is overwritten
  }

  // write T as [q][e] into qT_lds (overwriting the dead Q tile).
  // D-tile: M=e is register-contiguous -> one 16B store per tile per lane.
#pragma unroll
  for (int mt = 0; mt < 2; ++mt) {
    int ebase = (mtile0 + mt) * 16 + hif * 8;
#pragma unroll
    for (int nt = 0; nt < 4; ++nt) {
      int q = nt * 16 + ln;
      bf16x8 pk;
#pragma unroll
      for (int r = 0; r < 8; ++r) pk[r] = (__bf16)acc[mt][nt][r];
      *(bf16x8*)&qT_lds[q * QP + ebase] = pk;
    }
  }
  __syncthreads();

  // =======================================================================
  // Phase 2 (transposed): outT[a][q] = sum_e da[a][e] * T[q][e]
  //   A = da chunk [a][e] (M = a), B = T [q][e] (N = q)
  // =======================================================================
  v8f acc2[2][4];
#pragma unroll
  for (int mt = 0; mt < 2; ++mt)
#pragma unroll
    for (int nt = 0; nt < 4; ++nt) acc2[mt][nt] = vzero;

  const float* dab = da_em + (size_t)b * Adim * Ddim;

  for (int ec = 0; ec < Ddim; ec += DC) {
    // stage da[a][ec..ec+31] into chunk_lds[a][e]
    {
      const float4* src = (const float4*)(dab + (size_t)tid * Ddim + ec);
#pragma unroll
      for (int i = 0; i < DC / 8; ++i) {   // 4 iters
        float4 v0 = src[2 * i];
        float4 v1 = src[2 * i + 1];
        *(bf16x8*)&chunk_lds[tid * CP + i * 8] = cvt8(v0, v1);
      }
    }
    __syncthreads();

    v16bf afrag[2];
#pragma unroll
    for (int mt = 0; mt < 2; ++mt) {
      int a = (mtile0 + mt) * 16 + ln;
      afrag[mt] = mkfrag(&chunk_lds[a * CP + hif * 8],
                         &chunk_lds[a * CP + hif * 8 + 16]);
    }
    v16bf bfrag[4];
#pragma unroll
    for (int nt = 0; nt < 4; ++nt) {
      int q = nt * 16 + ln;
      bfrag[nt] = mkfrag(&qT_lds[q * QP + ec + hif * 8],
                         &qT_lds[q * QP + ec + hif * 8 + 16]);
    }
#pragma unroll
    for (int mt = 0; mt < 2; ++mt)
#pragma unroll
      for (int nt = 0; nt < 4; ++nt)
        acc2[mt][nt] = __builtin_amdgcn_wmma_f32_16x16x32_bf16(
            false, afrag[mt], false, bfrag[nt], (short)0, acc2[mt][nt], false, false);
    __syncthreads();
  }

  // ---- epilogue: + vq + vd + bias, sigmoid, vectorized float4 stores ----
  const float bk = bias[k];
  float* outp = out + (((size_t)b * Kdim + k) * Qdim + qbase) * Adim;
#pragma unroll
  for (int mt = 0; mt < 2; ++mt) {
    int abase = (mtile0 + mt) * 16 + hif * 8;     // a is register-contiguous
    float4 vd0 = *(const float4*)&vd_s[abase];
    float4 vd1 = *(const float4*)&vd_s[abase + 4];
#pragma unroll
    for (int nt = 0; nt < 4; ++nt) {
      int qloc = nt * 16 + ln;
      float vq = vq_s[qloc] + bk;
      const v8f& a8 = acc2[mt][nt];
      float4 o0, o1;
      o0.x = sigmoidf(a8[0] + vq + vd0.x);
      o0.y = sigmoidf(a8[1] + vq + vd0.y);
      o0.z = sigmoidf(a8[2] + vq + vd0.z);
      o0.w = sigmoidf(a8[3] + vq + vd0.w);
      o1.x = sigmoidf(a8[4] + vq + vd1.x);
      o1.y = sigmoidf(a8[5] + vq + vd1.y);
      o1.z = sigmoidf(a8[6] + vq + vd1.z);
      o1.w = sigmoidf(a8[7] + vq + vd1.w);
      float* dst = outp + (size_t)qloc * Adim + abase;
      *(float4*)dst       = o0;
      *(float4*)(dst + 4) = o1;
    }
  }
}

extern "C" void kernel_launch(void* const* d_in, const int* in_sizes, int n_in,
                              void* d_out, int out_size, void* d_ws, size_t ws_size,
                              hipStream_t stream) {
  (void)in_sizes; (void)n_in; (void)out_size; (void)d_ws; (void)ws_size;
  const float* q_em  = (const float*)d_in[0];
  const float* da_em = (const float*)d_in[1];
  const float* w     = (const float*)d_in[2];
  const float* V     = (const float*)d_in[3];
  const float* bias  = (const float*)d_in[4];
  float* out = (float*)d_out;

  dim3 grid(Bdim * Kdim * (Qdim / QT));  // 4096 workgroups: (b, k, q-tile)
  dim3 block(256);                       // 8 wave32 waves
  ntn_fused_wmma<<<grid, block, 0, stream>>>(q_em, da_em, w, V, bias, out);
}
